// primary_encoderwithksentimentatt2_57148834840719
// MI455X (gfx1250) — matrix operand order repaired and checked by reading, not compiled, not run on hardware
//
#include <hip/hip_runtime.h>
#include <hip/hip_bf16.h>
#include <math.h>

// ---------------------------------------------------------------------------
// Types for CDNA5 WMMA (wave32): v_wmma_f32_16x16x32_bf16
// ---------------------------------------------------------------------------
typedef __attribute__((ext_vector_type(16))) __bf16 v16bf;
typedef __attribute__((ext_vector_type(8)))  __bf16 v8bf;
typedef __attribute__((ext_vector_type(8)))  float  v8f;

union FragBF {
    v16bf v;
    struct { v8bf lo, hi; } p;
};

__device__ __forceinline__ unsigned short f2bfbits(float f) {
    unsigned u = __builtin_bit_cast(unsigned, f);
    unsigned r = u + 0x7FFFu + ((u >> 16) & 1u);   // round-to-nearest-even
    return (unsigned short)(r >> 16);
}
__device__ __forceinline__ float bfbits2f(unsigned short h) {
    return __builtin_bit_cast(float, (unsigned)h << 16);
}

// ---------------------------------------------------------------------------
// K1: embedding gather + fp32 -> bf16 convert
// ---------------------------------------------------------------------------
__global__ __launch_bounds__(256) void gather_kernel(
    const int* __restrict__ text, const float* __restrict__ table,
    unsigned short* __restrict__ g, int D)
{
    const int row = blockIdx.x;
    const float* src = table + (size_t)text[row] * D;
    unsigned short* dst = g + (size_t)row * D;
    for (int d = threadIdx.x; d < D; d += blockDim.x)
        dst[d] = f2bfbits(src[d]);
}

// ---------------------------------------------------------------------------
// K0: weight transpose + convert: W (K x N fp32 row-major) -> Wt (Npad x K bf16)
// rows n >= N are zero-padded.
// ---------------------------------------------------------------------------
__global__ __launch_bounds__(256) void wconv_kernel(
    const float* __restrict__ W, unsigned short* __restrict__ Wt,
    int K, int N, int Npad)
{
    int idx = blockIdx.x * blockDim.x + threadIdx.x;
    if (idx >= Npad * K) return;
    int n = idx / K, k = idx - n * K;
    Wt[idx] = (n < N) ? f2bfbits(W[(size_t)k * N + n]) : (unsigned short)0;
}

// ---------------------------------------------------------------------------
// Tiled bf16 WMMA GEMM: C = act(A @ Bt^T + bias), optional row mask.
//   A : M x K bf16 (row-major)
//   Bt: N x K bf16 (i.e. B transposed; B frag = two 16B chunks per lane)
//   Tile: 128(M) x 64(N), K-step 32.  256 threads = 8 waves, each wave 32x32.
//   act: 0=none, 1=relu, 2=tanh.  rowmask: zero row if s==0 || mask[row]==0.
// ---------------------------------------------------------------------------
#define TM 128
#define TN 64
#define TK 32

__global__ __launch_bounds__(256) void gemm_bf16_wmma(
    const unsigned short* __restrict__ A,
    const unsigned short* __restrict__ Bt,
    const float* __restrict__ bias, int biasN,
    unsigned short* __restrict__ Cb,      // bf16 out (nullable)
    float* __restrict__ Cf,               // f32 out (nullable)
    int M, int N, int K, int act,
    const int* __restrict__ rowmask, int S)
{
    __shared__ __align__(32) unsigned short sA[TM * TK];  // 8 KB
    __shared__ __align__(32) unsigned short sB[TN * TK];  // 4 KB

    const int tid   = threadIdx.x;
    const int lane  = tid & 31;
    const int wave  = tid >> 5;
    const int m0    = blockIdx.x * TM;
    const int n0    = blockIdx.y * TN;
    const int waveM = (wave & 3) << 5;   // 0,32,64,96
    const int waveN = (wave >> 2) << 5;  // 0,32
    const int hl    = lane & 15;
    const int hi    = lane >> 4;

    v8f acc[2][2] = {};

    for (int k0 = 0; k0 < K; k0 += TK) {
        __syncthreads();
        // stage A tile: 128 rows x 32 halfs (each thread: one 32B chunk)
        {
            const int r = tid >> 1;
            const int c = (tid & 1) << 4;
            const uint4* src = (const uint4*)(A + (size_t)(m0 + r) * K + k0 + c);
            uint4* dst = (uint4*)(sA + r * TK + c);
            dst[0] = src[0];
            dst[1] = src[1];
        }
        // stage B tile: 64 rows x 32 halfs (threads 0..127)
        if (tid < 128) {
            const int r = tid >> 1;
            const int c = (tid & 1) << 4;
            const uint4* src = (const uint4*)(Bt + (size_t)(n0 + r) * K + k0 + c);
            uint4* dst = (uint4*)(sB + r * TK + c);
            dst[0] = src[0];
            dst[1] = src[1];
        }
        // prefetch next K tile into the cache hierarchy (global_prefetch_b8)
        if (k0 + TK < K) {
            __builtin_prefetch(A + (size_t)(m0 + (tid >> 1)) * K + k0 + TK, 0, 1);
            if (tid < 128)
                __builtin_prefetch(Bt + (size_t)(n0 + (tid >> 1)) * K + k0 + TK, 0, 1);
        }
        __syncthreads();

        // A fragments (16-bit A 16x32 layout: lanes0-15 K0..7|16..23, lanes16-31 K8..15|24..31)
        FragBF a[2];
#pragma unroll
        for (int i = 0; i < 2; ++i) {
            const unsigned short* base = sA + (waveM + i * 16 + hl) * TK;
            a[i].p.lo = *(const v8bf*)(base + hi * 8);
            a[i].p.hi = *(const v8bf*)(base + 16 + hi * 8);
        }
        // B fragments (lanes = N col; halfs: K hi*8+0..7 | 16+hi*8+0..7) -> two 16B loads
        FragBF b[2];
#pragma unroll
        for (int j = 0; j < 2; ++j) {
            const unsigned short* base = sB + (waveN + j * 16 + hl) * TK;
            b[j].p.lo = *(const v8bf*)(base + hi * 8);
            b[j].p.hi = *(const v8bf*)(base + 16 + hi * 8);
        }
#pragma unroll
        for (int i = 0; i < 2; ++i)
#pragma unroll
            for (int j = 0; j < 2; ++j)
                acc[i][j] = __builtin_amdgcn_wmma_f32_16x16x32_bf16(
                    false, a[i].v, false, b[j].v, (short)0, acc[i][j], false, false);
    }

    // epilogue: bias + activation + mask + store (C 16x16 f32 layout:
    // VGPR v -> row hi*8+v, col = lane&15)
#pragma unroll
    for (int i = 0; i < 2; ++i) {
#pragma unroll
        for (int j = 0; j < 2; ++j) {
#pragma unroll
            for (int v = 0; v < 8; ++v) {
                const int row = m0 + waveM + i * 16 + hi * 8 + v;
                const int col = n0 + waveN + j * 16 + hl;
                float x = acc[i][j][v];
                if (bias && col < biasN) x += bias[col];
                if (act == 1)      x = fmaxf(x, 0.0f);
                else if (act == 2) x = tanhf(x);
                if (rowmask) {
                    const int s = row % S;
                    if (s == 0 || rowmask[row] == 0) x = 0.0f;
                }
                const size_t idx = (size_t)row * N + col;
                if (Cb) Cb[idx] = f2bfbits(x);
                if (Cf) Cf[idx] = x;
            }
        }
    }
}

// ---------------------------------------------------------------------------
// K4: attention score per (b,s):  a = T[row,0:100] . Ws + bs
// ---------------------------------------------------------------------------
__global__ __launch_bounds__(256) void score_kernel(
    const float* __restrict__ T, const float* __restrict__ Ws,
    const float* __restrict__ bs, float* __restrict__ score, int total)
{
    int r = blockIdx.x * blockDim.x + threadIdx.x;
    if (r >= total) return;
    const float* row = T + (size_t)r * 128;
    float acc = 0.0f;
    for (int j = 0; j < 100; ++j) acc += row[j] * Ws[j];
    score[r] = acc + bs[0];
}

// ---------------------------------------------------------------------------
// K5: softmax over s = 1..S-1 per batch (in place). attn[s=0] := 0
// ---------------------------------------------------------------------------
__global__ __launch_bounds__(256) void softmax_kernel(float* __restrict__ score, int S)
{
    __shared__ float red[256];
    const int b = blockIdx.x, t = threadIdx.x;
    float* a = score + (size_t)b * S;

    float mx = -1e30f;
    for (int s = 1 + t; s < S; s += 256) mx = fmaxf(mx, a[s]);
    red[t] = mx; __syncthreads();
    for (int off = 128; off; off >>= 1) {
        if (t < off) red[t] = fmaxf(red[t], red[t + off]);
        __syncthreads();
    }
    mx = red[0]; __syncthreads();

    float sm = 0.0f;
    for (int s = 1 + t; s < S; s += 256) {
        float e = __expf(a[s] - mx);
        a[s] = e;
        sm += e;
    }
    red[t] = sm; __syncthreads();
    for (int off = 128; off; off >>= 1) {
        if (t < off) red[t] += red[t + off];
        __syncthreads();
    }
    const float inv = 1.0f / red[0];
    for (int s = 1 + t; s < S; s += 256) a[s] *= inv;
    if (t == 0) a[0] = 0.0f;
}

// ---------------------------------------------------------------------------
// K6: avg[b,d] = sum_{s=1..S-1} attn[b,s] * hid[b,s,d]   (R = 1.0)
// ---------------------------------------------------------------------------
__global__ __launch_bounds__(256) void avg_kernel(
    const float* __restrict__ attn, const unsigned short* __restrict__ hid,
    float* __restrict__ avg, int S, int D)
{
    const int d = blockIdx.x * blockDim.x + threadIdx.x;
    const int b = blockIdx.y;
    float acc = 0.0f;
    const size_t base = (size_t)b * S;
    for (int s = 1; s < S; ++s)
        acc += attn[base + s] * bfbits2f(hid[(base + s) * D + d]);
    avg[(size_t)b * D + d] = acc;
}

// ---------------------------------------------------------------------------
// K7: heads — know/lcl projections, logits, l2-normalized supcon output
// ---------------------------------------------------------------------------
__global__ __launch_bounds__(256) void head_kernel(
    const float* __restrict__ avg, const float* __restrict__ last_hidden,
    const float* __restrict__ Wp1, const float* __restrict__ bp1,
    const float* __restrict__ Wp2, const float* __restrict__ bp2,
    const float* __restrict__ Wl,  const float* __restrict__ bl,
    float* __restrict__ out, int Bn, int S, int D, int E)
{
    __shared__ float s_avg[768], s_cls[768], s_know[768], s_lcl[768];
    __shared__ float red[256];
    __shared__ float s_ncls, s_navg;
    const int b = blockIdx.x, t = threadIdx.x;

    for (int d = t; d < D; d += 256) {
        s_avg[d] = avg[(size_t)b * D + d];
        s_cls[d] = last_hidden[(size_t)b * S * D + d];   // s = 0
    }
    __syncthreads();

    for (int d = t; d < D; d += 256) {
        float a1 = bp1[d], a2 = bp2[d];
        for (int k = 0; k < D; ++k) {
            a1 += s_avg[k] * Wp1[(size_t)k * D + d];
            a2 += s_cls[k] * Wp2[(size_t)k * D + d];
        }
        s_know[d] = fmaxf(a1, 0.0f);
        s_lcl[d]  = fmaxf(a2, 0.0f);
    }
    __syncthreads();

    if (t < E) {
        float acc = bl[t];
        for (int k = 0; k < D; ++k) acc += s_lcl[k]  * Wl[(size_t)k * E + t];
        for (int k = 0; k < D; ++k) acc += s_know[k] * Wl[(size_t)(D + k) * E + t];
        out[(size_t)b * E + t] = acc;
    }

    // L2 norms of cls0 and avg
    float p = 0.0f;
    for (int d = t; d < D; d += 256) p += s_cls[d] * s_cls[d];
    red[t] = p; __syncthreads();
    for (int off = 128; off; off >>= 1) { if (t < off) red[t] += red[t + off]; __syncthreads(); }
    if (t == 0) s_ncls = red[0];
    __syncthreads();

    p = 0.0f;
    for (int d = t; d < D; d += 256) p += s_avg[d] * s_avg[d];
    red[t] = p; __syncthreads();
    for (int off = 128; off; off >>= 1) { if (t < off) red[t] += red[t + off]; __syncthreads(); }
    if (t == 0) s_navg = red[0];
    __syncthreads();

    const float icls = 1.0f / fmaxf(sqrtf(s_ncls), 1e-12f);
    const float iavg = 1.0f / fmaxf(sqrtf(s_navg), 1e-12f);
    float* sup = out + (size_t)Bn * E;
    for (int d = t; d < D; d += 256) {
        sup[(size_t)b * 2 * D + d]     = s_cls[d] * icls;
        sup[(size_t)b * 2 * D + D + d] = s_avg[d] * iavg;
    }
}

// ---------------------------------------------------------------------------
// Host launch
// ---------------------------------------------------------------------------
extern "C" void kernel_launch(void* const* d_in, const int* in_sizes, int n_in,
                              void* d_out, int out_size, void* d_ws, size_t ws_size,
                              hipStream_t stream) {
    const int B = 64, S = 512, D = 768, H1 = 512, E = 7;
    const int M = B * S;

    const int*   text     = (const int*)  d_in[0];
    const int*   dictmask = (const int*)  d_in[1];
    const float* last_hid = (const float*)d_in[2];
    const float* table    = (const float*)d_in[3];
    const float* W1  = (const float*)d_in[4];
    const float* b1  = (const float*)d_in[5];
    const float* W2  = (const float*)d_in[6];
    const float* b2  = (const float*)d_in[7];
    const float* Wf  = (const float*)d_in[8];
    const float* bf  = (const float*)d_in[9];
    const float* Ws  = (const float*)d_in[10];
    const float* bs  = (const float*)d_in[11];
    const float* Wp1 = (const float*)d_in[12];
    const float* bp1 = (const float*)d_in[13];
    const float* Wp2 = (const float*)d_in[14];
    const float* bp2 = (const float*)d_in[15];
    const float* Wl  = (const float*)d_in[16];
    const float* bl  = (const float*)d_in[17];

    // carve workspace (256B aligned)
    char* base = (char*)d_ws;
    size_t off = 0;
    auto carve = [&](size_t bytes) -> void* {
        off = (off + 255) & ~(size_t)255;
        void* p = base + off;
        off += bytes;
        return p;
    };
    unsigned short* g    = (unsigned short*)carve((size_t)M * D  * 2);
    unsigned short* W1t  = (unsigned short*)carve((size_t)H1 * D * 2);
    unsigned short* W2t  = (unsigned short*)carve((size_t)D * H1 * 2);
    unsigned short* Wft  = (unsigned short*)carve((size_t)128 * D * 2);
    unsigned short* h    = (unsigned short*)carve((size_t)M * H1 * 2);
    unsigned short* hid  = (unsigned short*)carve((size_t)M * D  * 2);
    float*          T    = (float*)carve((size_t)M * 128 * 4);
    float*          sc   = (float*)carve((size_t)M * 4);
    float*          avg  = (float*)carve((size_t)B * D * 4);

    // gather + conversions
    gather_kernel<<<M, 256, 0, stream>>>(text, table, g, D);
    wconv_kernel<<<(H1 * D + 255) / 256, 256, 0, stream>>>(W1, W1t, D, H1, H1);
    wconv_kernel<<<(D * H1 + 255) / 256, 256, 0, stream>>>(W2, W2t, H1, D, D);
    wconv_kernel<<<(128 * D + 255) / 256, 256, 0, stream>>>(Wf, Wft, D, 100, 128);

    // GEMM1: h = relu(g @ W1 + b1)            (M x H1, K = D)
    gemm_bf16_wmma<<<dim3(M / TM, H1 / TN), 256, 0, stream>>>(
        g, W1t, b1, H1, h, nullptr, M, H1, D, /*act=*/1, nullptr, S);

    // GEMM2: hid = mask * relu(h @ W2 + b2)   (M x D, K = H1)
    gemm_bf16_wmma<<<dim3(M / TM, D / TN), 256, 0, stream>>>(
        h, W2t, b2, D, hid, nullptr, M, D, H1, /*act=*/1, dictmask, S);

    // GEMM3: T = tanh(hid @ Wf + bf)          (M x 128, K = D; cols 100..127 pad)
    gemm_bf16_wmma<<<dim3(M / TM, 128 / TN), 256, 0, stream>>>(
        hid, Wft, bf, 100, nullptr, T, M, 128, D, /*act=*/2, nullptr, S);

    // attention scores, softmax, weighted sum
    score_kernel<<<(M + 255) / 256, 256, 0, stream>>>(T, Ws, bs, sc, M);
    softmax_kernel<<<B, 256, 0, stream>>>(sc, S);
    avg_kernel<<<dim3(D / 256, B), 256, 0, stream>>>(sc, hid, avg, S, D);

    // heads + outputs
    head_kernel<<<B, 256, 0, stream>>>(avg, last_hid, Wp1, bp1, Wp2, bp2, Wl, bl,
                                       (float*)d_out, B, S, D, E);
}
